// EncoderLayer_77309411416
// MI455X (gfx1250) — matrix operand order repaired and checked by reading, not compiled, not run on hardware
//
#include <hip/hip_runtime.h>

typedef _Float16 f16;
typedef __attribute__((ext_vector_type(16))) _Float16 v16h;
typedef __attribute__((ext_vector_type(8)))  float    v8f;

#define D_MODEL 1024
#define D_INNER 4096
#define N_HEAD  16
#define DK      64
#define SEQ     2048
#define BATCH   2
#define ROWS    (BATCH * SEQ)   /* 4096 */
#define LN_EPS  1e-5f

#define WMMA_F16(a, b, c) \
  __builtin_amdgcn_wmma_f32_16x16x32_f16(false, (a), false, (b), (short)0, (c), false, false)

// ---------------------------------------------------------------------------
// WMMA fragment loaders (wave32; contiguous k order in A and B: slot s of lane half h holds k = 16h + s)
// ---------------------------------------------------------------------------

// A 16x32 (MxK) f16, row-major source with leading dim lda.
// lanes 0-15: M=lane, slots = K{0..15}; lanes 16-31: M=lane-16, K{16..31}
__device__ __forceinline__ v16h load_a_f16(const f16* __restrict__ A, int lda, int m0, int k0) {
  int lane = threadIdx.x & 31;
  const f16* p = A + (size_t)(m0 + (lane & 15)) * lda + k0 + ((lane >> 4) << 4);
  v16h a;
#pragma unroll
  for (int i = 0; i < 8; ++i) a[i] = p[i];
#pragma unroll
  for (int i = 0; i < 8; ++i) a[8 + i] = p[8 + i];
  return a;
}

// Same A layout but converting from an fp32 row-major source (attention probs).
__device__ __forceinline__ v16h load_a_f32(const float* __restrict__ A, int lda, int m0, int k0) {
  int lane = threadIdx.x & 31;
  const float* p = A + (size_t)(m0 + (lane & 15)) * lda + k0 + ((lane >> 4) << 4);
  v16h a;
#pragma unroll
  for (int i = 0; i < 8; ++i) a[i] = (f16)p[i];
#pragma unroll
  for (int i = 0; i < 8; ++i) a[8 + i] = (f16)p[8 + i];
  return a;
}

// B 32x16 (KxN) where B[k][n] = W[n][k], W row-major [N,K] (torch Linear y = x@W.T).
// lane = N (mod 16); half-wave selects K 0..15 / 16..31; K contiguous along W rows.
__device__ __forceinline__ v16h load_b_wT(const f16* __restrict__ W, int ldw, int k0, int n0) {
  int lane = threadIdx.x & 31;
  const f16* p = W + (size_t)(n0 + (lane & 15)) * ldw + k0 + ((lane >> 4) << 4);
  v16h b;
#pragma unroll
  for (int i = 0; i < 16; ++i) b[i] = p[i];
  return b;
}

// B 32x16 (KxN) from a row-major [K,N] source (V matrix): strided by ldv per K,
// but consecutive lanes hit consecutive addresses per slot (coalesced).
__device__ __forceinline__ v16h load_b_rm(const f16* __restrict__ V, int ldv, int k0, int n0) {
  int lane = threadIdx.x & 31;
  const f16* p = V + (size_t)(k0 + ((lane >> 4) << 4)) * ldv + n0 + (lane & 15);
  v16h b;
#pragma unroll
  for (int i = 0; i < 16; ++i) b[i] = p[(size_t)i * ldv];
  return b;
}

// ---------------------------------------------------------------------------
// Generic GEMM: C[M,N] = Xh[M,K] @ W[N,K]^T + bias.
// 128 threads = 4 waves; wave tile 32x64 (2 A frags x 4 B frags -> 8 WMMAs,
// 12 halves loaded per WMMA); block tile 128x64. Double-buffered K pipeline:
// next k-step's 6 fragments are in flight while current 8 WMMAs issue.
// Requires K % 32 == 0 and K >= 64 (true for 1024 / 4096).
// ---------------------------------------------------------------------------
__global__ void gemm_xwT(const f16* __restrict__ X, const f16* __restrict__ W,
                         const float* __restrict__ bias,
                         float* __restrict__ outF, f16* __restrict__ outH,
                         float hscale, int relu, int N, int K) {
  int wave = threadIdx.x >> 5, lane = threadIdx.x & 31;
  int m0 = blockIdx.y * 128 + wave * 32;
  int n0 = blockIdx.x * 64;
  v8f acc[2][4] = {};

  v16h a0 = load_a_f16(X, K, m0, 0);
  v16h a1 = load_a_f16(X, K, m0 + 16, 0);
  v16h b0 = load_b_wT(W, K, 0, n0);
  v16h b1 = load_b_wT(W, K, 0, n0 + 16);
  v16h b2 = load_b_wT(W, K, 0, n0 + 32);
  v16h b3 = load_b_wT(W, K, 0, n0 + 48);

  for (int k0 = 32; k0 < K; k0 += 32) {
    v16h na0 = load_a_f16(X, K, m0, k0);
    v16h na1 = load_a_f16(X, K, m0 + 16, k0);
    v16h nb0 = load_b_wT(W, K, k0, n0);
    v16h nb1 = load_b_wT(W, K, k0, n0 + 16);
    v16h nb2 = load_b_wT(W, K, k0, n0 + 32);
    v16h nb3 = load_b_wT(W, K, k0, n0 + 48);
    acc[0][0] = WMMA_F16(a0, b0, acc[0][0]);
    acc[1][0] = WMMA_F16(a1, b0, acc[1][0]);
    acc[0][1] = WMMA_F16(a0, b1, acc[0][1]);
    acc[1][1] = WMMA_F16(a1, b1, acc[1][1]);
    acc[0][2] = WMMA_F16(a0, b2, acc[0][2]);
    acc[1][2] = WMMA_F16(a1, b2, acc[1][2]);
    acc[0][3] = WMMA_F16(a0, b3, acc[0][3]);
    acc[1][3] = WMMA_F16(a1, b3, acc[1][3]);
    a0 = na0; a1 = na1; b0 = nb0; b1 = nb1; b2 = nb2; b3 = nb3;
  }
  acc[0][0] = WMMA_F16(a0, b0, acc[0][0]);
  acc[1][0] = WMMA_F16(a1, b0, acc[1][0]);
  acc[0][1] = WMMA_F16(a0, b1, acc[0][1]);
  acc[1][1] = WMMA_F16(a1, b1, acc[1][1]);
  acc[0][2] = WMMA_F16(a0, b2, acc[0][2]);
  acc[1][2] = WMMA_F16(a1, b2, acc[1][2]);
  acc[0][3] = WMMA_F16(a0, b3, acc[0][3]);
  acc[1][3] = WMMA_F16(a1, b3, acc[1][3]);

  int grp = lane >> 4, nn = lane & 15;
#pragma unroll
  for (int u = 0; u < 2; ++u)
#pragma unroll
    for (int t = 0; t < 4; ++t) {
      int n = n0 + t * 16 + nn;
      float bv = bias[n];
#pragma unroll
      for (int r = 0; r < 8; ++r) {
        int m = m0 + u * 16 + r + 8 * grp;   // C/D layout: VGPR r -> M=r / r+8
        float v = acc[u][t][r] + bv;
        if (relu) v = fmaxf(v, 0.f);
        if (outF) outF[(size_t)m * N + n] = v;
        if (outH) outH[(size_t)m * N + n] = (f16)(v * hscale);
      }
    }
}

// ---------------------------------------------------------------------------
// Attention scores: S_bh = Qh_bh @ Kh_bh^T (Q pre-scaled by 1/sqrt(dk)).
// blockIdx.z = bh = head*BATCH + batch (matches attn_flat layout).
// K-dim = 64 (two k-steps, fully unrolled); wave tile 32x64.
// ---------------------------------------------------------------------------
__global__ void attn_scores(const f16* __restrict__ Q, const f16* __restrict__ Kh,
                            float* __restrict__ attn) {
  int bh = blockIdx.z;
  int head = bh >> 1, batch = bh & 1;
  const f16* Qb = Q + (size_t)batch * SEQ * D_MODEL + head * DK;
  const f16* Kb = Kh + (size_t)batch * SEQ * D_MODEL + head * DK;
  int wave = threadIdx.x >> 5, lane = threadIdx.x & 31;
  int m0 = blockIdx.y * 128 + wave * 32;
  int n0 = blockIdx.x * 64;
  v8f acc[2][4] = {};
#pragma unroll
  for (int k0 = 0; k0 < DK; k0 += 32) {
    v16h a0 = load_a_f16(Qb, D_MODEL, m0, k0);
    v16h a1 = load_a_f16(Qb, D_MODEL, m0 + 16, k0);
#pragma unroll
    for (int t = 0; t < 4; ++t) {
      v16h b = load_b_wT(Kb, D_MODEL, k0, n0 + t * 16);
      acc[0][t] = WMMA_F16(a0, b, acc[0][t]);
      acc[1][t] = WMMA_F16(a1, b, acc[1][t]);
    }
  }
  float* out = attn + (size_t)bh * SEQ * SEQ;
  int grp = lane >> 4, nn = lane & 15;
#pragma unroll
  for (int u = 0; u < 2; ++u)
#pragma unroll
    for (int t = 0; t < 4; ++t)
#pragma unroll
      for (int r = 0; r < 8; ++r)
        out[(size_t)(m0 + u * 16 + r + 8 * grp) * SEQ + (n0 + t * 16 + nn)] = acc[u][t][r];
}

// ---------------------------------------------------------------------------
// Row softmax, in place over attn region of d_out: 65536 rows x 2048.
// ---------------------------------------------------------------------------
__global__ void softmax_rows(float* __restrict__ attn) {
  __shared__ float red[256];
  int tid = threadIdx.x;
  float* p = attn + (size_t)blockIdx.x * SEQ;
  float loc[8];
  float mx = -3.4e38f;
#pragma unroll
  for (int i = 0; i < 8; ++i) { loc[i] = p[tid + i * 256]; mx = fmaxf(mx, loc[i]); }
  red[tid] = mx; __syncthreads();
  for (int s = 128; s > 0; s >>= 1) { if (tid < s) red[tid] = fmaxf(red[tid], red[tid + s]); __syncthreads(); }
  mx = red[0]; __syncthreads();
  float sum = 0.f;
#pragma unroll
  for (int i = 0; i < 8; ++i) { loc[i] = __expf(loc[i] - mx); sum += loc[i]; }
  red[tid] = sum; __syncthreads();
  for (int s = 128; s > 0; s >>= 1) { if (tid < s) red[tid] += red[tid + s]; __syncthreads(); }
  float inv = 1.f / red[0];
#pragma unroll
  for (int i = 0; i < 8; ++i) p[tid + i * 256] = loc[i] * inv;
}

// ---------------------------------------------------------------------------
// ctx_bh[2048,64] = P_bh[2048,2048] @ V_bh[2048,64]; probs read fp32 from d_out,
// converted to f16 in registers. Wave tile 32x64; double-buffered A (the fat
// stream: 2048-deep K); B re-fetched per k-step (hits L2/WGP$).
// ---------------------------------------------------------------------------
__global__ void attn_ctx(const float* __restrict__ attn, const f16* __restrict__ V,
                         f16* __restrict__ ctx) {
  int bh = blockIdx.z;
  int head = bh >> 1, batch = bh & 1;
  const float* P = attn + (size_t)bh * SEQ * SEQ;
  const f16* Vb = V + (size_t)batch * SEQ * D_MODEL + head * DK;
  f16* Cb = ctx + (size_t)batch * SEQ * D_MODEL + head * DK;
  int wave = threadIdx.x >> 5, lane = threadIdx.x & 31;
  int m0 = blockIdx.y * 128 + wave * 32;
  v8f acc[2][4] = {};

  v16h a0 = load_a_f32(P, SEQ, m0, 0);
  v16h a1 = load_a_f32(P, SEQ, m0 + 16, 0);
  for (int k0 = 32; k0 < SEQ; k0 += 32) {
    v16h na0 = load_a_f32(P, SEQ, m0, k0);
    v16h na1 = load_a_f32(P, SEQ, m0 + 16, k0);
#pragma unroll
    for (int t = 0; t < 4; ++t) {
      v16h b = load_b_rm(Vb, D_MODEL, k0 - 32, t * 16);
      acc[0][t] = WMMA_F16(a0, b, acc[0][t]);
      acc[1][t] = WMMA_F16(a1, b, acc[1][t]);
    }
    a0 = na0; a1 = na1;
  }
#pragma unroll
  for (int t = 0; t < 4; ++t) {
    v16h b = load_b_rm(Vb, D_MODEL, SEQ - 32, t * 16);
    acc[0][t] = WMMA_F16(a0, b, acc[0][t]);
    acc[1][t] = WMMA_F16(a1, b, acc[1][t]);
  }

  int grp = lane >> 4, nn = lane & 15;
#pragma unroll
  for (int u = 0; u < 2; ++u)
#pragma unroll
    for (int t = 0; t < 4; ++t)
#pragma unroll
      for (int r = 0; r < 8; ++r)
        Cb[(size_t)(m0 + u * 16 + r + 8 * grp) * D_MODEL + t * 16 + nn] = (f16)acc[u][t][r];
}

// ---------------------------------------------------------------------------
// t = y + r; LayerNorm over 1024 cols; fp32 out (+ optional f16 copy).
// One 256-thread block per row, 4 cols per thread.
// ---------------------------------------------------------------------------
__global__ void residual_ln(const float* __restrict__ y, const float* __restrict__ r,
                            const float* __restrict__ g, const float* __restrict__ b,
                            float* __restrict__ outF, f16* __restrict__ outH) {
  __shared__ float red[256];
  int tid = threadIdx.x;
  size_t base = (size_t)blockIdx.x * D_MODEL;
  float loc[4];
  float s = 0.f;
#pragma unroll
  for (int i = 0; i < 4; ++i) {
    int c = tid + i * 256;
    loc[i] = y[base + c] + r[base + c];
    s += loc[i];
  }
  red[tid] = s; __syncthreads();
  for (int st = 128; st > 0; st >>= 1) { if (tid < st) red[tid] += red[tid + st]; __syncthreads(); }
  float mu = red[0] * (1.f / D_MODEL); __syncthreads();
  float s2 = 0.f;
#pragma unroll
  for (int i = 0; i < 4; ++i) { float d = loc[i] - mu; s2 += d * d; }
  red[tid] = s2; __syncthreads();
  for (int st = 128; st > 0; st >>= 1) { if (tid < st) red[tid] += red[tid + st]; __syncthreads(); }
  float rstd = rsqrtf(red[0] * (1.f / D_MODEL) + LN_EPS);
#pragma unroll
  for (int i = 0; i < 4; ++i) {
    int c = tid + i * 256;
    float o = (loc[i] - mu) * rstd * g[c] + b[c];
    outF[base + c] = o;
    if (outH) outH[base + c] = (f16)o;
  }
}

__global__ void cvt_f32_f16(const float* __restrict__ in, f16* __restrict__ out, int n) {
  int i = blockIdx.x * blockDim.x + threadIdx.x;
  if (i < n) out[i] = (f16)in[i];
}

// ---------------------------------------------------------------------------
// Host-side launch
// ---------------------------------------------------------------------------
extern "C" void kernel_launch(void* const* d_in, const int* in_sizes, int n_in,
                              void* d_out, int out_size, void* d_ws, size_t ws_size,
                              hipStream_t stream) {
  const float* x     = (const float*)d_in[0];
  const float* wq_w  = (const float*)d_in[1];
  const float* wq_b  = (const float*)d_in[2];
  const float* wk_w  = (const float*)d_in[3];
  const float* wk_b  = (const float*)d_in[4];
  const float* wv_w  = (const float*)d_in[5];
  const float* wv_b  = (const float*)d_in[6];
  const float* fc_w  = (const float*)d_in[7];
  const float* fc_b  = (const float*)d_in[8];
  const float* ln1_g = (const float*)d_in[9];
  const float* ln1_b = (const float*)d_in[10];
  const float* w1_w  = (const float*)d_in[11];
  const float* w1_b  = (const float*)d_in[12];
  const float* w2_w  = (const float*)d_in[13];
  const float* w2_b  = (const float*)d_in[14];
  const float* ln2_g = (const float*)d_in[15];
  const float* ln2_b = (const float*)d_in[16];

  float* enc_out = (float*)d_out;                                   // [4096,1024]
  float* attn    = enc_out + (size_t)ROWS * D_MODEL;                // [32,2048,2048]

  // workspace layout; hid aliases the q/k/v/ctx region (dead by then)
  const size_t MB = 1024 * 1024;
  char* w = (char*)d_ws;
  f16* wqh   = (f16*)(w + 0 * MB);    // 2 MiB each
  f16* wkh   = (f16*)(w + 2 * MB);
  f16* wvh   = (f16*)(w + 4 * MB);
  f16* fch   = (f16*)(w + 6 * MB);
  f16* w1h   = (f16*)(w + 8 * MB);    // 8 MiB
  f16* w2h   = (f16*)(w + 16 * MB);   // 8 MiB
  f16* xh    = (f16*)(w + 24 * MB);   // 8 MiB
  f16* qh    = (f16*)(w + 32 * MB);   // 8 MiB, aliased by hid later
  f16* kh    = (f16*)(w + 40 * MB);   // 8 MiB, aliased by hid later
  f16* vh    = (f16*)(w + 48 * MB);   // 8 MiB, aliased by hid later
  f16* ctxh  = (f16*)(w + 56 * MB);   // 8 MiB, aliased by hid later
  f16* hid   = (f16*)(w + 32 * MB);   // 32 MiB, written after ctx/q/k/v consumed
  float* ao  = (float*)(w + 64 * MB); // 16 MiB fp32 post-LN1 (residual for LN2)
  f16* aoh   = (f16*)(w + 80 * MB);   // 8 MiB  f16 post-LN1 (w1 input)
  float* tmp = (float*)(w + 88 * MB); // 16 MiB fp32 GEMM results (fc, then w2)

  const int NW = D_MODEL * D_MODEL;       // 1M weight elements
  const int NX = ROWS * D_MODEL;          // 4M activation elements
  dim3 b256(256);

  // 1) fp32 -> f16 converts
  cvt_f32_f16<<<dim3((NX + 255) / 256), b256, 0, stream>>>(x, xh, NX);
  cvt_f32_f16<<<dim3((NW + 255) / 256), b256, 0, stream>>>(wq_w, wqh, NW);
  cvt_f32_f16<<<dim3((NW + 255) / 256), b256, 0, stream>>>(wk_w, wkh, NW);
  cvt_f32_f16<<<dim3((NW + 255) / 256), b256, 0, stream>>>(wv_w, wvh, NW);
  cvt_f32_f16<<<dim3((NW + 255) / 256), b256, 0, stream>>>(fc_w, fch, NW);
  cvt_f32_f16<<<dim3((D_INNER * D_MODEL + 255) / 256), b256, 0, stream>>>(w1_w, w1h, D_INNER * D_MODEL);
  cvt_f32_f16<<<dim3((D_INNER * D_MODEL + 255) / 256), b256, 0, stream>>>(w2_w, w2h, D_INNER * D_MODEL);

  dim3 gblk(128);
  // 2) Q/K/V projections (Q pre-scaled by 1/sqrt(64) = 0.125 before f16 cast)
  dim3 gQKV(D_MODEL / 64, ROWS / 128);
  gemm_xwT<<<gQKV, gblk, 0, stream>>>(xh, wqh, wq_b, nullptr, qh, 0.125f, 0, D_MODEL, D_MODEL);
  gemm_xwT<<<gQKV, gblk, 0, stream>>>(xh, wkh, wk_b, nullptr, kh, 1.0f,   0, D_MODEL, D_MODEL);
  gemm_xwT<<<gQKV, gblk, 0, stream>>>(xh, wvh, wv_b, nullptr, vh, 1.0f,   0, D_MODEL, D_MODEL);

  // 3) scores -> d_out attn region (raw), 4) softmax in place
  attn_scores<<<dim3(SEQ / 64, SEQ / 128, N_HEAD * BATCH), gblk, 0, stream>>>(qh, kh, attn);
  softmax_rows<<<dim3(N_HEAD * BATCH * SEQ), b256, 0, stream>>>(attn);

  // 5) ctx = P @ V
  attn_ctx<<<dim3(1, SEQ / 128, N_HEAD * BATCH), gblk, 0, stream>>>(attn, vh, ctxh);

  // 6) fc projection, then residual + LN1
  gemm_xwT<<<gQKV, gblk, 0, stream>>>(ctxh, fch, fc_b, tmp, nullptr, 1.0f, 0, D_MODEL, D_MODEL);
  residual_ln<<<dim3(ROWS), b256, 0, stream>>>(tmp, x, ln1_g, ln1_b, ao, aoh);

  // 7) FFN: w1 + ReLU (f16), w2 (fp32), residual + LN2 -> enc_output
  gemm_xwT<<<dim3(D_INNER / 64, ROWS / 128), gblk, 0, stream>>>(aoh, w1h, w1_b, nullptr, hid, 1.0f, 1, D_INNER, D_MODEL);
  gemm_xwT<<<gQKV, gblk, 0, stream>>>(hid, w2h, w2_b, tmp, nullptr, 1.0f, 0, D_MODEL, D_INNER);
  residual_ln<<<dim3(ROWS), b256, 0, stream>>>(tmp, ao, ln2_g, ln2_b, enc_out, nullptr);
}
